// LocalSpectralTokenizer_69011534512768
// MI455X (gfx1250) — matrix operand order repaired and checked by reading, not compiled, not run on hardware
//
#include <hip/hip_runtime.h>
#include <hip/hip_bf16.h>
#include <math.h>

// ---------------- problem constants (from reference) ----------------
#define BATCH   32
#define NCH     64
#define LEN     8192
#define NW      1021          // (LEN-32)/8 + 1 sliding windows, start = 8*w
#define NMAX    2048          // hard upper bound on tokens (<= L/8 fulls + L/8 tails)
#define PMIN    8
#define PMAX_   64
#define PENALTY 3.0f

typedef __attribute__((ext_vector_type(16))) _Float16 v16h;
typedef __attribute__((ext_vector_type(8)))  float    v8f;

// ---------------- K0: zero the whole output ----------------
__global__ __launch_bounds__(256) void k_zero(float* out, int n) {
    for (int i = blockIdx.x * blockDim.x + threadIdx.x; i < n;
         i += gridDim.x * blockDim.x)
        out[i] = 0.0f;
}

// ---------------- K1: channel mean (bandwidth-bound pass #1) ----------------
__global__ __launch_bounds__(256) void k_mean(const float* __restrict__ x,
                                              float* __restrict__ agg) {
    int b = blockIdx.y;
    int t = blockIdx.x * 256 + threadIdx.x;           // LEN/256 blocks in x
    const float* xb = x + (size_t)b * NCH * LEN;
    float s = 0.0f;
    #pragma unroll 4
    for (int c = 0; c < NCH; ++c) {
        int cn = (c + 1 < NCH) ? c + 1 : c;
        __builtin_prefetch(&xb[(size_t)cn * LEN + t], 0, 0);  // global_prefetch_b8
        s += xb[(size_t)c * LEN + t];
    }
    agg[(size_t)b * LEN + t] = s * (1.0f / (float)NCH);
}

// ---------------- K2: sliding 32-pt DFT power spectrum via WMMA ----------------
// One wave computes 16 windows x 16 bins with two v_wmma_f32_16x16x32_f16
// (cos plane, sin plane), then p = re^2 + im^2, then per-window features.
__global__ __launch_bounds__(128) void k_spectral(const float* __restrict__ agg,
                                                  float* __restrict__ feat) {
    const int b    = blockIdx.x;
    const int wave = threadIdx.x >> 5;
    const int lane = threadIdx.x & 31;
    const int wbase = (blockIdx.y * 4 + wave) * 16;   // first window of this wave
    const float* a = agg + (size_t)b * LEN;

    const int col   = lane & 15;        // N index -> frequency bin col+1
    const int bin   = col + 1;
    const int khalf = (lane < 16) ? 0 : 16;

    // B fragments (32x16, col-striped): element e holds K = khalf + e
    v16h Bc, Bs;
    #pragma unroll
    for (int e = 0; e < 16; ++e) {
        int k = khalf + e;
        float ang = -6.28318530717958647692f * (float)(bin * k) * (1.0f / 32.0f);
        Bc[e] = (_Float16)__cosf(ang);
        Bs[e] = (_Float16)__sinf(ang);
    }

    // A fragment (16x32): row = lane&15 -> window wbase+row
    const int row = lane & 15;
    const int w   = wbase + row;
    const int st  = (w < NW ? w : NW - 1) * 8;        // clamp, keep EXEC uniform
    const float scale = (w < NW) ? 1.0f : 0.0f;       // zero out-of-range windows
    v16h A;
    #pragma unroll
    for (int e = 0; e < 16; ++e) {
        int k = e + ((lane < 16) ? 0 : 16);
        A[e] = (_Float16)(a[st + k] * scale);
    }

    v8f zacc = {};
    v8f re = __builtin_amdgcn_wmma_f32_16x16x32_f16(false, A, false, Bc,
                                                    (short)0, zacc, false, false);
    v8f im = __builtin_amdgcn_wmma_f32_16x16x32_f16(false, A, false, Bs,
                                                    (short)0, zacc, false, false);

    __shared__ float lds_p[4][16][17];                // [wave][window][bin], padded
    #pragma unroll
    for (int r = 0; r < 8; ++r) {
        int m = r + ((lane < 16) ? 0 : 8);
        lds_p[wave][m][col] = re[r] * re[r] + im[r] * im[r];
    }
    __syncthreads();

    if (lane < 16) {                                   // one window per lane
        int w2 = wbase + lane;
        if (w2 < NW) {
            float total = 0.0f, sf = 0.0f, sf2 = 0.0f;
            #pragma unroll
            for (int k = 0; k < 16; ++k) {
                float p = lds_p[wave][lane][k];
                float f = (float)(k + 1) * (1.0f / 32.0f);
                total += p; sf += p * f; sf2 += p * f * f;
            }
            float inv = 1.0f / (total + 1e-12f);
            float cen = sf * inv;
            float var = sf2 * inv - cen * cen;
            float spread = sqrtf(fmaxf(var, 0.0f));
            float* fo = feat + ((size_t)b * NW + w2) * 3;
            fo[0] = cen;
            fo[1] = 2.0f * spread;
            fo[2] = log1pf(total * (1.0f / 32.0f));
        }
    }
}

// ---------------- K3: PELT DP changepoints (one block per batch) ----------------
__global__ __launch_bounds__(256) void k_dp(const float* __restrict__ feat,
                                            int* __restrict__ parent_out) {
    const int b = blockIdx.x;
    const int tid = threadIdx.x;
    __shared__ float cs[NW + 1][3];
    __shared__ float cs2[NW + 1][3];
    __shared__ float F[NW + 1];
    __shared__ int   par[NW + 1];
    __shared__ float rmin[256];
    __shared__ int   rarg[256];

    if (tid == 0) {
        for (int d = 0; d < 3; ++d) { cs[0][d] = 0.0f; cs2[0][d] = 0.0f; }
        for (int t = 1; t <= NW; ++t)
            for (int d = 0; d < 3; ++d) {
                float v = feat[((size_t)b * NW + (t - 1)) * 3 + d];
                cs[t][d]  = cs[t - 1][d]  + v;
                cs2[t][d] = cs2[t - 1][d] + v * v;
            }
        F[0] = 0.0f;
    }
    __syncthreads();

    for (int t = 1; t <= NW; ++t) {
        float best = INFINITY; int barg = 0;
        for (int j = tid; j < t; j += 256) {
            float inv = 1.0f / (float)(t - j);
            float cost = 0.0f;
            #pragma unroll
            for (int d = 0; d < 3; ++d) {
                float d1 = cs[t][d]  - cs[j][d];
                float d2 = cs2[t][d] - cs2[j][d];
                cost += d2 - d1 * d1 * inv;
            }
            float v = F[j] + cost + PENALTY;
            if (v < best) { best = v; barg = j; }
        }
        rmin[tid] = best; rarg[tid] = barg;
        __syncthreads();
        for (int s = 128; s > 0; s >>= 1) {
            if (tid < s) {
                float o = rmin[tid + s]; int oa = rarg[tid + s];
                if (o < rmin[tid] || (o == rmin[tid] && oa < rarg[tid])) {
                    rmin[tid] = o; rarg[tid] = oa;
                }
            }
            __syncthreads();
        }
        if (tid == 0) { F[t] = rmin[0]; par[t] = rarg[0]; }
        __syncthreads();
    }
    for (int t = tid; t <= NW; t += 256)
        parent_out[(size_t)b * (NW + 1) + t] = par[t];
}

// ---------------- K4: backtrack, segment DFT stats, token plan ----------------
__global__ __launch_bounds__(256) void k_plan(const float* __restrict__ agg,
                                              const int* __restrict__ parent,
                                              int* __restrict__ tokS,
                                              int* __restrict__ tokE,
                                              int* __restrict__ ntok,
                                              float* out_mask, float* out_so,
                                              float* out_eo, float* out_co,
                                              float* out_sp, float* out_rg,
                                              float* out_nt, int N) {
    const int b = blockIdx.x;
    const int tid = threadIdx.x;
    const float* a = agg + (size_t)b * LEN;

    __shared__ int chain[NW + 2];
    __shared__ int bnds[NW + 3];
    __shared__ int nb;
    __shared__ float r_tot[256], r_sf[256], r_sf2[256], r_pm[256];
    __shared__ int   r_pa[256];

    if (tid == 0) {
        int cnt = 0, t = NW;
        while (t > 0) { chain[cnt++] = t; t = parent[(size_t)b * (NW + 1) + t]; }
        chain[cnt++] = 0;
        int nv = 0;
        bnds[nv++] = 0;
        for (int i = cnt - 2; i >= 1; --i) {          // interior CPs, ascending
            int w = chain[i];
            if (w < 0 || w >= NW) continue;
            int ti = 8 * w;                            // starts[w] = 8*w
            int last = bnds[nv - 1];
            if (ti <= last) continue;
            if (ti - last < PMIN) continue;
            if (LEN - ti < PMIN) continue;
            bnds[nv++] = ti;
        }
        bnds[nv++] = LEN;
        nb = nv;
    }
    __syncthreads();

    const int nseg = nb - 1;
    int tcount = 0;                                    // meaningful on tid==0
    for (int si = 0; si < nseg; ++si) {
        const int s = bnds[si], e = bnds[si + 1];
        const int Ls = e - s;
        const int K  = Ls / 2;                         // bins 1..K (DC dropped)

        float total = 0.0f, sf = 0.0f, sf2 = 0.0f, pmax = -1.0f;
        int parg = 0;
        for (int k = tid + 1; k <= K; k += 256) {
            float th = 6.28318530717958647692f * (float)k / (float)Ls;
            float cd = __cosf(th), sd = __sinf(th);
            float re = 0.0f, im = 0.0f, c = 1.0f, sn = 0.0f;
            for (int n0 = 0; n0 < Ls; ++n0) {          // phase recurrence
                float v = a[s + n0];
                re += v * c; im += v * sn;
                float cn = c * cd - sn * sd;
                sn = sn * cd + c * sd;
                c  = cn;
            }
            float p = re * re + im * im;
            float f = (float)k / (float)Ls;
            total += p; sf += p * f; sf2 += p * f * f;
            if (p > pmax) { pmax = p; parg = k - 1; }
        }
        r_tot[tid] = total; r_sf[tid] = sf; r_sf2[tid] = sf2;
        r_pm[tid] = pmax;   r_pa[tid] = parg;
        __syncthreads();
        for (int st2 = 128; st2 > 0; st2 >>= 1) {
            if (tid < st2) {
                r_tot[tid] += r_tot[tid + st2];
                r_sf[tid]  += r_sf[tid + st2];
                r_sf2[tid] += r_sf2[tid + st2];
                float o = r_pm[tid + st2]; int oa = r_pa[tid + st2];
                if (o > r_pm[tid] || (o == r_pm[tid] && oa < r_pa[tid])) {
                    r_pm[tid] = o; r_pa[tid] = oa;
                }
            }
            __syncthreads();
        }

        if (tid == 0) {
            float tot = r_tot[0];
            float inv = 1.0f / (tot + 1e-12f);
            float cen = r_sf[0] * inv;
            float var = r_sf2[0] * inv - cen * cen;
            float bw  = 2.0f * sqrtf(fmaxf(var, 0.0f));
            float dp  = (float)Ls / ((float)r_pa[0] + 1.0f);

            float raw = dp / (1.0f + bw);
            int pl = (int)rintf(raw * 0.5f) * 2;
            pl = pl < PMIN ? PMIN : (pl > PMAX_ ? PMAX_ : pl);

            float rg0 = dp * (1.0f / (float)LEN);
            float rg2 = (float)Ls * (1.0f / (float)LEN);
            const float dc = (float)(LEN - 1), dsz = (float)LEN;

            int n_full = Ls / pl;
            for (int i = 0; i <= n_full; ++i) {
                int ts, te;
                if (i < n_full)      { ts = s + i * pl; te = ts + pl; }       // 'full'
                else {                                                         // 'tail'
                    if (n_full * pl >= Ls) break;
                    ts = s + n_full * pl; te = e;
                }
                if (tcount < NMAX) {
                    tokS[(size_t)b * NMAX + tcount] = ts;
                    tokE[(size_t)b * NMAX + tcount] = te;
                    if (tcount < N) {
                        size_t o = (size_t)b * N + tcount;
                        out_mask[o] = 1.0f;
                        out_so[o]   = (float)ts;
                        out_eo[o]   = (float)te;
                        out_co[o]   = ((float)ts + (float)te - 1.0f) * 0.5f / dc;
                        out_sp[o]   = (float)(te - ts) / dsz;
                        out_rg[o * 3 + 0] = rg0;
                        out_rg[o * 3 + 1] = bw;
                        out_rg[o * 3 + 2] = rg2;
                    }
                    ++tcount;
                }
            }
        }
        __syncthreads();                               // reuse reduction arrays
    }
    if (tid == 0) { ntok[b] = tcount; out_nt[b] = (float)tcount; }
}

// ---------------- K5: patch gather + linear resize (pass #2 over x) ----------------
__global__ __launch_bounds__(256) void k_patch(const float* __restrict__ x,
                                               const int* __restrict__ tokS,
                                               const int* __restrict__ tokE,
                                               const int* __restrict__ ntok,
                                               float* __restrict__ out_patch,
                                               int N) {
    const int bt  = blockIdx.x;
    const int b   = bt / N;
    const int tok = bt % N;
    const int valid = (tok < ntok[b]) ? 1 : 0;
    const int s = valid ? tokS[(size_t)b * NMAX + tok] : 0;
    const int e = valid ? tokE[(size_t)b * NMAX + tok] : 1;
    const int P = e - s;
    const float scale = (float)P * (1.0f / 16.0f);
    float* ob = out_patch + (size_t)bt * NCH * 16;

    for (int idx = threadIdx.x; idx < NCH * 16; idx += 256) {
        int c = idx >> 4, j = idx & 15;
        float pos = ((float)j + 0.5f) * scale - 0.5f;
        pos = fminf(fmaxf(pos, 0.0f), (float)(P - 1));
        int i0 = (int)floorf(pos);
        int i1 = i0 + 1 < P ? i0 + 1 : P - 1;
        float wt = pos - (float)i0;
        const float* xc = x + ((size_t)b * NCH + c) * LEN + s;
        float v = xc[i0] * (1.0f - wt) + xc[i1] * wt;
        ob[idx] = valid ? v : 0.0f;
    }
}

// ---------------- launcher ----------------
extern "C" void kernel_launch(void* const* d_in, const int* in_sizes, int n_in,
                              void* d_out, int out_size, void* d_ws, size_t ws_size,
                              hipStream_t stream) {
    const float* x = (const float*)d_in[0];
    float* out = (float*)d_out;

    // out_size = B*(N*1032 + 1)  ->  recover data-dependent N deterministically
    long long per_b = (long long)out_size / BATCH;
    int N = (int)((per_b - 1) / 1032);
    if (N < 1) N = 1;

    // workspace carve-up (~2.2 MB; fits comfortably in 192 MB L2)
    char* wp = (char*)d_ws;
    float* agg  = (float*)wp; wp += (size_t)BATCH * LEN * sizeof(float);
    float* feat = (float*)wp; wp += (size_t)BATCH * NW * 3 * sizeof(float);
    int* parent = (int*)wp;   wp += (size_t)BATCH * (NW + 1) * sizeof(int);
    int* tokS   = (int*)wp;   wp += (size_t)BATCH * NMAX * sizeof(int);
    int* tokE   = (int*)wp;   wp += (size_t)BATCH * NMAX * sizeof(int);
    int* ntok   = (int*)wp;

    // output regions (reference return order, flattened as float)
    size_t o = 0;
    float* out_patch = out + o; o += (size_t)BATCH * N * NCH * 16;
    float* out_mask  = out + o; o += (size_t)BATCH * N;
    float* out_so    = out + o; o += (size_t)BATCH * N;
    float* out_eo    = out + o; o += (size_t)BATCH * N;
    float* out_co    = out + o; o += (size_t)BATCH * N;
    float* out_sp    = out + o; o += (size_t)BATCH * N;
    float* out_rg    = out + o; o += (size_t)BATCH * N * 3;
    float* out_nt    = out + o;

    int zgrid = (out_size + 1023) / 1024; if (zgrid < 1) zgrid = 1;
    k_zero<<<zgrid, 256, 0, stream>>>(out, out_size);
    k_mean<<<dim3(LEN / 256, BATCH), 256, 0, stream>>>(x, agg);
    k_spectral<<<dim3(BATCH, 16), 128, 0, stream>>>(agg, feat);   // 16*64 >= 1021 windows
    k_dp<<<BATCH, 256, 0, stream>>>(feat, parent);
    k_plan<<<BATCH, 256, 0, stream>>>(agg, parent, tokS, tokE, ntok,
                                      out_mask, out_so, out_eo, out_co,
                                      out_sp, out_rg, out_nt, N);
    k_patch<<<BATCH * N, 256, 0, stream>>>(x, tokS, tokE, ntok, out_patch, N);
}
